// CSTreeLSTM_37117107372689
// MI455X (gfx1250) — compile-verified, matrix operand order checked by reading, not run on hardware
//
#include <hip/hip_runtime.h>

// ---------------------------------------------------------------------------
// CDNA5 (gfx1250) TreeLSTM: level-batched bf16 WMMA GEMMs + fused LSTM cell.
// Block = 8 wave32s arranged 4(M) x 2(N): block tile = 128 nodes x 32 cols.
// Each wave computes two 16-node M-subtiles x 16 cols, sharing B fragments;
// B is further shared across the 4 M-waves and A across the 2 N-waves via
// WGP$, minimizing L2 traffic per output (~(M + 4N)*K amortization).
// ---------------------------------------------------------------------------

typedef __attribute__((ext_vector_type(16))) __bf16 v16bf;
typedef __attribute__((ext_vector_type(8)))  float  v8f;
typedef __attribute__((ext_vector_type(8)))  int    v8i;
typedef __attribute__((ext_vector_type(4)))  int    v4i;

#define TDEPTH 15
#define NNODES ((1 << TDEPTH) - 1)   // 32767
#define DDIM   512
#define HDIM   512
#define DK     (DDIM + HDIM)         // 1024 = K of the concat GEMM
#define MBLK   128                   // nodes per block (4 M-waves x 32)

// ---- fragment helpers ------------------------------------------------------

// Load a 16x32 bf16 A-fragment (or B^T-fragment) from a row-major matrix.
// ISA layout (CDNA5 ISA): lanes 0-15 hold row M=lane, K in {0..7, 16..23};
// lanes 16-31 hold row M=lane-16, K in {8..15, 24..31}. Two 16B loads/lane.
static __device__ __forceinline__ v8i load_frag(const __bf16* __restrict__ base,
                                                int ldk, int lane) {
    const int m  = lane & 15;
    const int ko = (lane & 16) ? 8 : 0;
    const __bf16* p = base + (size_t)m * (size_t)ldk + ko;
    v4i lo = *(const v4i*)(p);
    v4i hi = *(const v4i*)(p + 16);
    v8i r;
    r[0] = lo[0]; r[1] = lo[1]; r[2] = lo[2]; r[3] = lo[3];
    r[4] = hi[0]; r[5] = hi[1]; r[6] = hi[2]; r[7] = hi[3];
    return r;
}

static __device__ __forceinline__ v8f wmma_bf16(v8i a, v8i b, v8f c) {
    return __builtin_amdgcn_wmma_f32_16x16x32_bf16(
        /*neg_a=*/false, __builtin_bit_cast(v16bf, a),
        /*neg_b=*/false, __builtin_bit_cast(v16bf, b),
        /*c_mod=*/(short)0, c, /*reuse_a=*/false, /*reuse_b=*/false);
}

// Packed bf16 add of two fragments: h_sum = h_l + h_r (v_pk_add_bf16).
static __device__ __forceinline__ v8i pk_add_bf16(v8i a, v8i b) {
    v8i d;
#pragma unroll
    for (int i = 0; i < 8; ++i) {
        int t;
        asm("v_pk_add_bf16 %0, %1, %2" : "=v"(t) : "v"(a[i]), "v"(b[i]));
        d[i] = t;
    }
    return d;
}

static __device__ __forceinline__ float sigm(float x) {
    return 1.0f / (1.0f + __expf(-x));
}
static __device__ __forceinline__ float tanh_fast(float x) {
    float e = __expf(2.0f * x);
    return 1.0f - 2.0f / (e + 1.0f);   // well-behaved for |x| large
}

// ---- conversion kernel -----------------------------------------------------

__global__ void cvt_f32_bf16(const float* __restrict__ in,
                             __bf16* __restrict__ out, int n) {
    for (int i = blockIdx.x * blockDim.x + threadIdx.x; i < n;
         i += gridDim.x * blockDim.x)
        out[i] = (__bf16)in[i];
}

// ---- level kernel ----------------------------------------------------------
// One launch per tree level. wbf holds [Wi | Wf | Wo | Wu], each HDIM x DK
// row-major bf16.
template <bool LEAF>
__global__ __launch_bounds__(256)
void tree_level_kernel(const __bf16* __restrict__ xbf,
                       __bf16* __restrict__ hbf,
                       float* __restrict__ cfp,
                       const __bf16* __restrict__ wbf,
                       const float* __restrict__ bi,
                       const float* __restrict__ bff,
                       const float* __restrict__ bo,
                       const float* __restrict__ bu,
                       float* __restrict__ out,
                       int start, int cnt, int is_root) {
    const int lane = threadIdx.x & 31;
    const int wv   = threadIdx.x >> 5;
    const int wm   = wv >> 1;                    // 0..3  (M direction)
    const int wn   = wv & 1;                     // 0..1  (N direction)
    const int m0   = blockIdx.x * MBLK + wm * 32;   // per-wave node base
    const int n0   = blockIdx.y * 32 + wn * 16;     // per-wave column base
    const int col  = n0 + (lane & 15);

    const __bf16* Wi = wbf + (size_t)0 * HDIM * DK;
    const __bf16* Wf = wbf + (size_t)1 * HDIM * DK;
    const __bf16* Wo = wbf + (size_t)2 * HDIM * DK;
    const __bf16* Wu = wbf + (size_t)3 * HDIM * DK;

    v8f acc_i[2]  = {{}, {}}, acc_o[2] = {{}, {}}, acc_u[2] = {{}, {}};
    v8f acc_fs[2] = {{}, {}}, acc_fl[2] = {{}, {}}, acc_fr[2] = {{}, {}};

    // ---- x half: K = 0..DDIM, A = x rows (shared by all gates) ----
    const __bf16* Ax0 = xbf + (size_t)(start + m0) * DDIM;
    const __bf16* Ax1 = Ax0 + (size_t)16 * DDIM;
    for (int k = 0; k < DDIM; k += 32) {
        v8i a0 = load_frag(Ax0 + k, DDIM, lane);
        v8i a1 = load_frag(Ax1 + k, DDIM, lane);
        v8i b0 = load_frag(Wi + (size_t)n0 * DK + k, DK, lane);
        acc_i[0] = wmma_bf16(a0, b0, acc_i[0]);
        acc_i[1] = wmma_bf16(a1, b0, acc_i[1]);
        v8i b1 = load_frag(Wo + (size_t)n0 * DK + k, DK, lane);
        acc_o[0] = wmma_bf16(a0, b1, acc_o[0]);
        acc_o[1] = wmma_bf16(a1, b1, acc_o[1]);
        v8i b2 = load_frag(Wu + (size_t)n0 * DK + k, DK, lane);
        acc_u[0] = wmma_bf16(a0, b2, acc_u[0]);
        acc_u[1] = wmma_bf16(a1, b2, acc_u[1]);
        if constexpr (!LEAF) {
            v8i b3 = load_frag(Wf + (size_t)n0 * DK + k, DK, lane);
            acc_fs[0] = wmma_bf16(a0, b3, acc_fs[0]);  // shared x-half of f_l/f_r
            acc_fs[1] = wmma_bf16(a1, b3, acc_fs[1]);
        }
    }

    // ---- h half: K = DDIM..DK, A = children h rows (row stride 2*HDIM) ----
    if constexpr (!LEAF) {
        // child rows of node (start+m0+m): left = 2*(start+m0+m)+1
        const __bf16* Ahl0 = hbf + (size_t)(2 * (start + m0) + 1) * HDIM;
        const __bf16* Ahr0 = Ahl0 + HDIM;
        const __bf16* Ahl1 = Ahl0 + (size_t)32 * HDIM;   // +16 nodes = +32 rows
        const __bf16* Ahr1 = Ahl1 + HDIM;
        for (int k = 0; k < HDIM; k += 32) {
            v8i al0 = load_frag(Ahl0 + k, 2 * HDIM, lane);
            v8i ar0 = load_frag(Ahr0 + k, 2 * HDIM, lane);
            v8i as0 = pk_add_bf16(al0, ar0);    // h_l + h_r  (subtile 0)
            v8i al1 = load_frag(Ahl1 + k, 2 * HDIM, lane);
            v8i ar1 = load_frag(Ahr1 + k, 2 * HDIM, lane);
            v8i as1 = pk_add_bf16(al1, ar1);    // h_l + h_r  (subtile 1)
            const int kk = DDIM + k;
            v8i b0 = load_frag(Wi + (size_t)n0 * DK + kk, DK, lane);
            acc_i[0] = wmma_bf16(as0, b0, acc_i[0]);
            acc_i[1] = wmma_bf16(as1, b0, acc_i[1]);
            v8i b1 = load_frag(Wo + (size_t)n0 * DK + kk, DK, lane);
            acc_o[0] = wmma_bf16(as0, b1, acc_o[0]);
            acc_o[1] = wmma_bf16(as1, b1, acc_o[1]);
            v8i b2 = load_frag(Wu + (size_t)n0 * DK + kk, DK, lane);
            acc_u[0] = wmma_bf16(as0, b2, acc_u[0]);
            acc_u[1] = wmma_bf16(as1, b2, acc_u[1]);
            v8i b3 = load_frag(Wf + (size_t)n0 * DK + kk, DK, lane);
            acc_fl[0] = wmma_bf16(al0, b3, acc_fl[0]);
            acc_fr[0] = wmma_bf16(ar0, b3, acc_fr[0]);
            acc_fl[1] = wmma_bf16(al1, b3, acc_fl[1]);
            acc_fr[1] = wmma_bf16(ar1, b3, acc_fr[1]);
        }
    }

    // ---- fused LSTM cell epilogue ----
    const float Bi = bi[col], Bf = bff[col], Bo = bo[col], Bu = bu[col];
    const int mbase = (lane >> 4) * 8;          // C/D layout: M = r + 8*(lane/16)
#pragma unroll
    for (int s = 0; s < 2; ++s) {
#pragma unroll
        for (int r = 0; r < 8; ++r) {
            const int m    = s * 16 + r + mbase;
            const int node = start + m0 + m;
            const float iv = sigm(acc_i[s][r] + Bi);
            const float ov = sigm(acc_o[s][r] + Bo);
            const float uv = tanh_fast(acc_u[s][r] + Bu);
            float cv;
            if constexpr (LEAF) {
                cv = iv * uv;
            } else {
                const int gl = 2 * node + 1, gr = gl + 1;
                const float cL  = cfp[(size_t)gl * HDIM + col];
                const float cR  = cfp[(size_t)gr * HDIM + col];
                const float flv = sigm(acc_fs[s][r] + acc_fl[s][r] + Bf);
                const float frv = sigm(acc_fs[s][r] + acc_fr[s][r] + Bf);
                cv = iv * uv + flv * cL + frv * cR;
            }
            const float hv = ov * tanh_fast(cv);
            if (m0 + m < cnt) {
                cfp[(size_t)node * HDIM + col] = cv;
                hbf[(size_t)node * HDIM + col] = (__bf16)hv;
                if (!LEAF && is_root && node == 0) {
                    out[col]        = hv;   // h[0]
                    out[HDIM + col] = cv;   // c[0]
                }
            }
        }
    }
}

// ---------------------------------------------------------------------------

extern "C" void kernel_launch(void* const* d_in, const int* in_sizes, int n_in,
                              void* d_out, int out_size, void* d_ws, size_t ws_size,
                              hipStream_t stream) {
    const float* x  = (const float*)d_in[0];
    const float* Wi = (const float*)d_in[1];
    const float* bi = (const float*)d_in[2];
    const float* Wf = (const float*)d_in[3];
    const float* bf = (const float*)d_in[4];
    const float* Wo = (const float*)d_in[5];
    const float* bo = (const float*)d_in[6];
    const float* Wu = (const float*)d_in[7];
    const float* bu = (const float*)d_in[8];
    float* out = (float*)d_out;

    // Workspace layout (all 16B aligned):
    //   wbf : 4 * HDIM * DK           bf16  (4 MB)   [Wi|Wf|Wo|Wu]
    //   xbf : NNODES * DDIM           bf16  (32 MB)
    //   hbf : NNODES * HDIM           bf16  (32 MB)
    //   cfp : NNODES * HDIM           f32   (64 MB)
    __bf16* wbf = (__bf16*)d_ws;
    __bf16* xbf = wbf + (size_t)4 * HDIM * DK;
    __bf16* hbf = xbf + (size_t)NNODES * DDIM;
    float*  cfp = (float*)(hbf + (size_t)NNODES * HDIM);

    const int CVB = 256, CVG = 2048;
    cvt_f32_bf16<<<CVG, CVB, 0, stream>>>(Wi, wbf + (size_t)0 * HDIM * DK, HDIM * DK);
    cvt_f32_bf16<<<CVG, CVB, 0, stream>>>(Wf, wbf + (size_t)1 * HDIM * DK, HDIM * DK);
    cvt_f32_bf16<<<CVG, CVB, 0, stream>>>(Wo, wbf + (size_t)2 * HDIM * DK, HDIM * DK);
    cvt_f32_bf16<<<CVG, CVB, 0, stream>>>(Wu, wbf + (size_t)3 * HDIM * DK, HDIM * DK);
    cvt_f32_bf16<<<CVG, CVB, 0, stream>>>(x, xbf, NNODES * DDIM);

    // Levels leaves -> root; each launch is a batched GEMM + fused cell.
    for (int lvl = TDEPTH - 1; lvl >= 0; --lvl) {
        const int start = (1 << lvl) - 1;
        const int cnt   = (1 << lvl);
        const int mt    = (cnt + MBLK - 1) / MBLK;
        dim3 grid(mt, HDIM / 32);
        if (lvl == TDEPTH - 1) {
            tree_level_kernel<true><<<grid, 256, 0, stream>>>(
                xbf, hbf, cfp, wbf, bi, bf, bo, bu, out, start, cnt, 0);
        } else {
            tree_level_kernel<false><<<grid, 256, 0, stream>>>(
                xbf, hbf, cfp, wbf, bi, bf, bo, bu, out, start, cnt,
                (lvl == 0) ? 1 : 0);
        }
    }
}
